// SimpleGCN_21225728377320
// MI455X (gfx1250) — matrix operand order repaired and checked by reading, not compiled, not run on hardware
//
#include <hip/hip_runtime.h>
#include <hip/hip_bf16.h>

// ---------- types ----------
typedef __attribute__((ext_vector_type(16))) __bf16       bf16x16;
typedef __attribute__((ext_vector_type(8)))  float        f32x8;
typedef __attribute__((ext_vector_type(4)))  unsigned int u32x4;

union Frag32B {           // 32 bytes: a v16bf WMMA fragment, loadable as 2x b128
    bf16x16 v;
    u32x4   q[2];
};

// Pin a fragment: volatile asm = scheduling barrier ordered against all memory
// ops; "+v" redefines the registers so the consuming WMMA cannot hoist past it.
__device__ __forceinline__ void pin_frag(Frag32B& fr) {
    asm volatile("" : "+v"(fr.q[0]), "+v"(fr.q[1]));
}

__device__ __forceinline__ unsigned short f32_to_bf16_rne(float f) {
    unsigned int u = __float_as_uint(f);
    u += 0x7FFFu + ((u >> 16) & 1u);   // round to nearest even
    return (unsigned short)(u >> 16);
}

// ---------- prep kernels ----------
__global__ void cvt_f32_to_bf16(const float* __restrict__ in,
                                unsigned short* __restrict__ out, int n) {
    int i = blockIdx.x * blockDim.x + threadIdx.x;
    if (i < n) out[i] = f32_to_bf16_rne(in[i]);
}

// Pack W [K,N] f32 (row-major) into per-lane WMMA B fragments:
// Bp[((nt*Ks + ks)*32 + lane)*16 + e] = bf16(W[(ks*32 + (e>>3)*16 + (lane>>4)*8 + (e&7))*N + nt*16 + (lane&15)])
__global__ void pack_B_frags(const float* __restrict__ W,
                             unsigned short* __restrict__ Bp,
                             int K, int N, int Ntiles) {
    int idx = blockIdx.x * blockDim.x + threadIdx.x;
    int Ks = K >> 5;
    int total = Ntiles * Ks * 32 * 16;
    if (idx >= total) return;
    int e    = idx & 15;
    int lane = (idx >> 4) & 31;
    int ksnt = idx >> 9;
    int ks   = ksnt % Ks;
    int nt   = ksnt / Ks;
    int n = nt * 16 + (lane & 15);
    int k = ks * 32 + (e >> 3) * 16 + (lane >> 4) * 8 + (e & 7);
    float v = (n < N) ? W[(size_t)k * N + n] : 0.0f;
    Bp[idx] = f32_to_bf16_rne(v);
}

__global__ void deg_init(float* __restrict__ dinv, int n) {
    int i = blockIdx.x * blockDim.x + threadIdx.x;
    if (i < n) dinv[i] = 1.0f;                      // self-loop
}
__global__ void deg_count(const long long* __restrict__ dst,
                          float* __restrict__ dinv, int e) {
    int i = blockIdx.x * blockDim.x + threadIdx.x;
    if (i < e) atomicAdd(&dinv[(int)dst[i]], 1.0f);
}
__global__ void deg_rsqrt(float* __restrict__ dinv, int n) {
    int i = blockIdx.x * blockDim.x + threadIdx.x;
    if (i < n) dinv[i] = rsqrtf(dinv[i]);           // deg >= 1 always
}

// ---------- WMMA GEMM: C[M,Nout] = A[M,K](bf16) * Bpacked ----------
// MT=5 divides Mtiles=3125 exactly -> no guards. Flattened modulo-scheduled
// pipeline: A rotating buffer depth 5 / lookahead 4; B depth 2 / 1 K-step ahead.
// Fragment "pins" keep each WMMA barrier-ordered after 3+ later prefetch loads.
// Single base pointer per matrix; all displacements are constant IOFFSETs.
// grid = (Mtiles/5, ceil(Nout/16)), block = 32 (one wave per block)
template<int K, int Nout>
__global__ void gemm_wmma_bf16(const unsigned short* __restrict__ A,
                               const unsigned short* __restrict__ Bp,
                               float* __restrict__ C) {
    constexpr int MT    = 5;
    constexpr int Ks    = K / 32;
    constexpr int STEPS = Ks * MT;
    constexpr int LA    = 4;            // A lookahead (< buffer depth 5)

    const int lane = threadIdx.x;       // 0..31
    const int hi   = lane >> 4;         // 0/1
    const int l15  = lane & 15;
    const int nt   = blockIdx.y;
    const int mt0  = blockIdx.x * MT;

    // single per-lane base pointers
    const unsigned short* abase = A + (size_t)(mt0 * 16 + l15) * K + hi * 8;
    const unsigned short* bbase = Bp + ((size_t)nt * Ks * 32 + lane) * 16;

    f32x8   acc[MT] = {};
    Frag32B ab[5];                      // A rotating buffer
    Frag32B bb[2];                      // B rotating buffer

    auto loadA = [&](int f, Frag32B& dst) {
        // constant displacement: tile stride 16*K elems, K-step stride 32 elems
        const int off = (f % MT) * 16 * K + (f / MT) * 32;
        dst.q[0] = *(const u32x4*)(abase + off);        // e 0..7  -> K = hi*8 + e
        dst.q[1] = *(const u32x4*)(abase + off + 16);   // e 8..15 -> K = 16+hi*8+(e-8)
    };
    auto loadB = [&](int ks, Frag32B& dst) {
        const int off = ks * (32 * 16);
        dst.q[0] = *(const u32x4*)(bbase + off);
        dst.q[1] = *(const u32x4*)(bbase + off + 8);
    };

    // prologue
    loadB(0, bb[0]);
    #pragma unroll
    for (int i = 0; i < LA && i < STEPS; ++i) loadA(i, ab[i % 5]);

    #pragma unroll
    for (int f = 0; f < STEPS; ++f) {
        const int ks = f / MT;
        const int t  = f % MT;
        // pin operands: WMMA(f) is now ordered after all previously issued
        // prefetch loads (iterations f-3..f-1) -> counted waits, real overlap
        pin_frag(ab[f % 5]);
        if (t == 0) pin_frag(bb[ks & 1]);
        acc[t] = __builtin_amdgcn_wmma_f32_16x16x32_bf16(
                     false, ab[f % 5].v, false, bb[ks & 1].v,
                     (short)0, acc[t], false, false);
        // prefetch for future steps (issued after WMMA in program order)
        if (f + LA < STEPS) loadA(f + LA, ab[(f + LA) % 5]);
        if (t == 0 && ks + 1 < Ks) loadB(ks + 1, bb[(ks + 1) & 1]);
    }

    // store 16x16 tiles: VGPR r -> row r + 8*hi, col = nt*16 + (lane&15)
    const int col = nt * 16 + l15;
    if (col < Nout) {
        float* cbase = C + (size_t)(mt0 * 16 + 8 * hi) * Nout + col;
        #pragma unroll
        for (int t = 0; t < MT; ++t)
            #pragma unroll
            for (int r = 0; r < 8; ++r)
                cbase[(size_t)(t * 16 + r) * Nout] = acc[t][r];
    }
}

// ---------- edge aggregation: one wave32 per (edge | self-loop) ----------
template<int F>
__global__ void aggregate(const float* __restrict__ h,
                          const long long* __restrict__ src,
                          const long long* __restrict__ dst,
                          const float* __restrict__ dinv,
                          float* __restrict__ out,
                          int E, int N) {
    int gtid = blockIdx.x * blockDim.x + threadIdx.x;
    int wid  = gtid >> 5;
    int lane = gtid & 31;
    int total = E + N;
    if (wid >= total) return;                  // wave-uniform
    int s, d;
    if (wid < E) { s = (int)src[wid]; d = (int)dst[wid]; }
    else         { s = wid - E; d = s; }
    float norm = dinv[s] * dinv[d];
    const float* hs = h   + (size_t)s * F;
    float*       od = out + (size_t)d * F;
    __builtin_prefetch(hs + lane, 0, 1);       // global_prefetch into L2
    #pragma unroll
    for (int f = lane; f < F; f += 32)
        atomicAdd(&od[f], hs[f] * norm);       // global_atomic_add_f32 (L2-resident)
}

// ---------- bias + relu (2D indexing, no integer div) ----------
__global__ void bias_relu_to_bf16(const float* __restrict__ in,
                                  const float* __restrict__ bias,
                                  unsigned short* __restrict__ out, int F) {
    int col = blockIdx.x * blockDim.x + threadIdx.x;
    int row = blockIdx.y;
    if (col >= F) return;
    size_t i = (size_t)row * F + col;
    float v = in[i] + bias[col];
    out[i] = f32_to_bf16_rne(v > 0.0f ? v : 0.0f);
}
__global__ void bias_relu_inplace(float* __restrict__ io,
                                  const float* __restrict__ bias, int F) {
    int col = blockIdx.x * blockDim.x + threadIdx.x;
    int row = blockIdx.y;
    if (col >= F) return;
    size_t i = (size_t)row * F + col;
    float v = io[i] + bias[col];
    io[i] = v > 0.0f ? v : 0.0f;
}

// ---------- host ----------
static inline int cdiv(long long a, long long b) { return (int)((a + b - 1) / b); }

extern "C" void kernel_launch(void* const* d_in, const int* in_sizes, int n_in,
                              void* d_out, int out_size, void* d_ws, size_t ws_size,
                              hipStream_t stream) {
    const float*     x  = (const float*)d_in[0];      // [N,128]
    const long long* ei = (const long long*)d_in[1];  // [2,E] int64
    const float*     W1 = (const float*)d_in[2];      // [128,512]
    const float*     b1 = (const float*)d_in[3];      // [512]
    const float*     W2 = (const float*)d_in[4];      // [512,250]
    const float*     b2 = (const float*)d_in[5];      // [250]

    const int N  = in_sizes[0] / 128;   // 50000
    const int E  = in_sizes[1] / 2;     // 800000
    const int F0 = 128, F1 = 512, F2 = 250;
    const int N1t = F1 / 16;            // 32
    const int N2t = (F2 + 15) / 16;     // 16 (padded in Bp)
    const int Ks1 = F0 / 32, Ks2 = F1 / 32;
    const int Mtiles = N / 16;          // 3125 = 5^5

    const long long* srcp = ei;
    const long long* dstp = ei + E;

    // workspace carve-up (256-B aligned)
    char* base = (char*)d_ws;
    size_t off = 0;
    auto alloc = [&](size_t bytes) { char* p = base + off; off = (off + bytes + 255) & ~(size_t)255; return p; };
    float*          dinv = (float*)         alloc((size_t)N * 4);
    unsigned short* xb   = (unsigned short*)alloc((size_t)N * F0 * 2);
    unsigned short* W1p  = (unsigned short*)alloc((size_t)N1t * Ks1 * 32 * 16 * 2);
    unsigned short* W2p  = (unsigned short*)alloc((size_t)N2t * Ks2 * 32 * 16 * 2);
    float*          h1   = (float*)         alloc((size_t)N * F1 * 4);
    float*          agg1 = (float*)         alloc((size_t)N * F1 * 4);
    unsigned short* hb   = (unsigned short*)alloc((size_t)N * F1 * 2);
    float*          h2   = h1;  // h1 is dead after aggregation 1; reuse for GEMM2 output
    float*          out  = (float*)d_out;

    const int B = 256;

    // prep
    cvt_f32_to_bf16<<<cdiv((long long)N * F0, B), B, 0, stream>>>(x, xb, N * F0);
    pack_B_frags<<<cdiv((long long)N1t * Ks1 * 512, B), B, 0, stream>>>(W1, W1p, F0, F1, N1t);
    pack_B_frags<<<cdiv((long long)N2t * Ks2 * 512, B), B, 0, stream>>>(W2, W2p, F1, F2, N2t);
    deg_init <<<cdiv(N, B), B, 0, stream>>>(dinv, N);
    deg_count<<<cdiv(E, B), B, 0, stream>>>(dstp, dinv, E);
    deg_rsqrt<<<cdiv(N, B), B, 0, stream>>>(dinv, N);

    // layer 1: h1 = x @ W1 (bf16 WMMA, f32 acc)
    {
        dim3 g(Mtiles / 5, N1t);
        gemm_wmma_bf16<128, 512><<<g, 32, 0, stream>>>(xb, W1p, h1);
    }
    (void)hipMemsetAsync(agg1, 0, (size_t)N * F1 * 4, stream);
    aggregate<512><<<cdiv((long long)(E + N) * 32, B), B, 0, stream>>>(h1, srcp, dstp, dinv, agg1, E, N);
    {
        dim3 g(cdiv(F1, B), N);
        bias_relu_to_bf16<<<g, B, 0, stream>>>(agg1, b1, hb, F1);
    }

    // layer 2: h2 = relu1 @ W2
    {
        dim3 g(Mtiles / 5, N2t);
        gemm_wmma_bf16<512, 250><<<g, 32, 0, stream>>>(hb, W2p, h2);
    }
    (void)hipMemsetAsync(out, 0, (size_t)N * F2 * 4, stream);
    aggregate<250><<<cdiv((long long)(E + N) * 32, B), B, 0, stream>>>(h2, srcp, dstp, dinv, out, E, N);
    {
        dim3 g(cdiv(F2, B), N);
        bias_relu_inplace<<<g, B, 0, stream>>>(out, b2, F2);
    }
}
